// NetVLAD_71717363908944
// MI455X (gfx1250) — matrix operand order repaired and checked by reading, not compiled, not run on hardware
//
#include <hip/hip_runtime.h>
#include <math.h>

// Problem constants (fixed by setup_inputs)
#define N_ 32
#define C_ 512
#define P_ 1024   // H*W = 32*32
#define K_ 64

typedef __attribute__((ext_vector_type(16))) _Float16 v16h;
typedef __attribute__((ext_vector_type(8)))  _Float16 v8h;
typedef __attribute__((ext_vector_type(8)))  float    v8f;

// Build a 16-half WMMA operand from two 16-byte chunks (CDNA5 A/B layouts
// put K lo/hi halves 16 elements apart for the two lane groups).
__device__ __forceinline__ v16h ldpair(const _Float16* p0, const _Float16* p1) {
  v8h lo = *(const v8h*)p0;
  v8h hi = *(const v8h*)p1;
  v16h r;
#pragma unroll
  for (int i = 0; i < 8; ++i) { r[i] = lo[i]; r[i + 8] = hi[i]; }
  return r;
}

__device__ __forceinline__ v8f wmma16x16x32(v16h a, v16h b, v8f c) {
  // D = A(16x32 f16) * B(32x16 f16) + C(16x16 f32)
  return __builtin_amdgcn_wmma_f32_16x16x32_f16(
      /*neg_a=*/false, a, /*neg_b=*/false, b,
      /*c_mod=*/(short)0, c, /*reuse_a=*/false, /*reuse_b=*/false);
}

// ---------------------------------------------------------------------------
// Kernel 1: per-pixel L2 normalize over C; emit f16 in two layouts.
//   xn_cm[n][c][p]  (channel-major, B operand of vlad matmul)
//   xn_pm[n][p][c]  (pixel-major,   B operand of logits matmul)
// Also converts conv_w -> f16 (one element per thread; grid*block == K_*C_).
// ---------------------------------------------------------------------------
__global__ void __launch_bounds__(256)
normalize_kernel(const float* __restrict__ x, const float* __restrict__ conv_w,
                 _Float16* __restrict__ xn_cm, _Float16* __restrict__ xn_pm,
                 _Float16* __restrict__ wh) {
  const int tid = threadIdx.x;
  const int gt  = blockIdx.x * 256 + tid;
  if (gt < K_ * C_) wh[gt] = (_Float16)conv_w[gt];

  const int n = blockIdx.x >> 2;                 // 4 blocks per image
  const int p = (blockIdx.x & 3) * 256 + tid;    // pixel id
  const float* xp = x + (size_t)n * C_ * P_ + p; // stride P_ over c; coalesced over p
  float ss = 0.f;
#pragma unroll 4
  for (int c = 0; c < C_; ++c) { float v = xp[(size_t)c * P_]; ss += v * v; }
  const float scale = 1.f / fmaxf(sqrtf(ss), 1e-12f);
#pragma unroll 4
  for (int c = 0; c < C_; ++c) {
    _Float16 h = (_Float16)(xp[(size_t)c * P_] * scale);
    xn_cm[((size_t)n * C_ + c) * P_ + p] = h;    // coalesced
    xn_pm[((size_t)n * P_ + p) * C_ + c] = h;    // transposed copy
  }
}

// ---------------------------------------------------------------------------
// Kernel 2: logits via WMMA + softmax over K per pixel.
// Block = (n, 128-pixel tile). 8 waves; wave w owns pixel columns w*16..+15
// and accumulates all 4 k-tiles over 16 K-steps (C=512, 32 per step).
// Writes sa[n][k][p] (f16, row-major in p = vlad A operand) and per-block
// partial sums sasum_part[n][ptile][k] (deterministic, no atomics).
// ---------------------------------------------------------------------------
__global__ void __launch_bounds__(256)
logits_softmax_kernel(const _Float16* __restrict__ xn_pm,
                      const _Float16* __restrict__ wh,
                      const float* __restrict__ conv_b,
                      _Float16* __restrict__ sa,
                      float* __restrict__ sasum_part) {
  __shared__ float lg[128][K_ + 1];              // +1 pad kills bank conflicts
  const int tid = threadIdx.x;
  const int w   = tid >> 5;
  const int l   = tid & 31;
  const int ln  = l & 15;                        // lane-in-group
  const int lh  = l >> 4;                        // lane group (K lo/hi half)
  const int n   = blockIdx.x >> 3;
  const int pt  = blockIdx.x & 7;
  const int pb  = pt * 128;

  const int pcol = pb + w * 16 + ln;             // this lane's pixel (B column)
  const _Float16* xrow = xn_pm + ((size_t)n * P_ + pcol) * C_;

  v8f acc0 = {}, acc1 = {}, acc2 = {}, acc3 = {};
  for (int cb = 0; cb < C_; cb += 32) {
    // B: lane=pixel column, K(c) lo/hi chunks at cb+8*lh and cb+8*lh+16
    v16h b = ldpair(xrow + cb + 8 * lh, xrow + cb + 8 * lh + 16);
    const int c0 = cb + 8 * lh;                  // A K-chunk start per lane group
    v16h a0 = ldpair(wh + (0 * 16 + ln) * C_ + c0, wh + (0 * 16 + ln) * C_ + c0 + 16);
    v16h a1 = ldpair(wh + (1 * 16 + ln) * C_ + c0, wh + (1 * 16 + ln) * C_ + c0 + 16);
    v16h a2 = ldpair(wh + (2 * 16 + ln) * C_ + c0, wh + (2 * 16 + ln) * C_ + c0 + 16);
    v16h a3 = ldpair(wh + (3 * 16 + ln) * C_ + c0, wh + (3 * 16 + ln) * C_ + c0 + 16);
    acc0 = wmma16x16x32(a0, b, acc0);
    acc1 = wmma16x16x32(a1, b, acc1);
    acc2 = wmma16x16x32(a2, b, acc2);
    acc3 = wmma16x16x32(a3, b, acc3);
  }

  // Scatter D tiles (+bias) to LDS: VGPR r holds M=r (lanes 0-15) / M=r+8.
  const int plocal = w * 16 + ln;
#pragma unroll
  for (int r = 0; r < 8; ++r) {
    int kr = r + 8 * lh;
    lg[plocal][0 * 16 + kr] = acc0[r] + conv_b[0 * 16 + kr];
    lg[plocal][1 * 16 + kr] = acc1[r] + conv_b[1 * 16 + kr];
    lg[plocal][2 * 16 + kr] = acc2[r] + conv_b[2 * 16 + kr];
    lg[plocal][3 * 16 + kr] = acc3[r] + conv_b[3 * 16 + kr];
  }
  __syncthreads();

  // Stable softmax over K=64 per pixel; 128 threads = 128 pixels.
  if (tid < 128) {
    float m = -1e30f;
#pragma unroll
    for (int k = 0; k < K_; ++k) m = fmaxf(m, lg[tid][k]);
    float s = 0.f;
#pragma unroll
    for (int k = 0; k < K_; ++k) s += __expf(lg[tid][k] - m);
    const float inv = 1.f / s;
#pragma unroll
    for (int k = 0; k < K_; ++k) {
      float v = __expf(lg[tid][k] - m) * inv;
      lg[tid][k] = v;
      sa[((size_t)(n * K_ + k)) * P_ + pb + tid] = (_Float16)v;  // coalesced in tid
    }
  }
  __syncthreads();

  // Per-block column sums of sa over the 128 pixels (deterministic partials).
  if (tid < K_) {
    float s = 0.f;
    for (int p = 0; p < 128; ++p) s += lg[p][tid];
    sasum_part[((size_t)n * 8 + pt) * K_ + tid] = s;
  }
}

// ---------------------------------------------------------------------------
// Kernel 2b: fold the 8 per-tile partials into sasum[n][k].
// ---------------------------------------------------------------------------
__global__ void reduce_sasum_kernel(const float* __restrict__ part,
                                    float* __restrict__ sasum) {
  const int n = blockIdx.x, k = threadIdx.x;
  float s = 0.f;
#pragma unroll
  for (int t = 0; t < 8; ++t) s += part[((size_t)n * 8 + t) * K_ + k];
  sasum[n * K_ + k] = s;
}

// ---------------------------------------------------------------------------
// Kernel 3: vlad[n][k][c] = sum_p sa[n][k][p]*xn[n][c][p] - sasum[n][k]*cent[k][c]
// One wave per 16x16 (k,c) output tile; 32 WMMA steps over P=1024.
// Also emits per-(n,k,ctile) sum-of-squares partials (shfl reduce, no atomics).
// ---------------------------------------------------------------------------
__global__ void __launch_bounds__(256)
vlad_kernel(const _Float16* __restrict__ sa, const _Float16* __restrict__ xn_cm,
            const float* __restrict__ sasum, const float* __restrict__ cent,
            float* __restrict__ vlad, float* __restrict__ normpart) {
  const int tid = threadIdx.x;
  const int w   = tid >> 5;
  const int l   = tid & 31;
  const int ln  = l & 15;
  const int lh  = l >> 4;
  const int gw  = blockIdx.x * 8 + w;            // 0..4095 wave tiles
  const int n   = gw >> 7;
  const int kt  = (gw >> 5) & 3;                 // 4 k-tiles
  const int ct  = gw & 31;                       // 32 c-tiles

  const _Float16* srow = sa    + ((size_t)(n * K_ + kt * 16 + ln)) * P_; // A row k
  const _Float16* xcol = xn_cm + ((size_t)(n * C_ + ct * 16 + ln)) * P_; // B col c

  v8f acc = {};
  for (int pb = 0; pb < P_; pb += 32) {
    __builtin_prefetch(xcol + pb + 64, 0, 0);    // global_prefetch_b8 ahead of B
    const int p0 = pb + 8 * lh;
    v16h a = ldpair(srow + p0, srow + p0 + 16);          // A: K(p) lo/hi chunks
    v16h b = ldpair(xcol + p0, xcol + p0 + 16);          // B: K(p) lo/hi chunks
    acc = wmma16x16x32(a, b, acc);
  }

  const int c = ct * 16 + ln;
#pragma unroll
  for (int r = 0; r < 8; ++r) {
    const int k = kt * 16 + r + 8 * lh;
    float val = acc[r] - sasum[n * K_ + k] * cent[k * C_ + c];
    vlad[((size_t)(n * K_ + k)) * C_ + c] = val;
    // reduce val^2 over the 16 lanes sharing this k (lanes l and l^16 differ in k)
    float sq = val * val;
    sq += __shfl_xor(sq, 1);
    sq += __shfl_xor(sq, 2);
    sq += __shfl_xor(sq, 4);
    sq += __shfl_xor(sq, 8);
    if (ln == 0) normpart[((size_t)(n * K_ + k)) * 32 + ct] = sq;
  }
}

// ---------------------------------------------------------------------------
// Kernel 4: intra-normalize over C then global L2 over K*C.
//   sum_c (v/max(sqrt(s),eps))^2 == s / max(s, eps^2)  -> cheap global norm.
// ---------------------------------------------------------------------------
__global__ void __launch_bounds__(256)
finalize_kernel(const float* __restrict__ vlad, const float* __restrict__ normpart,
                float* __restrict__ out) {
  __shared__ float inv1[K_];
  __shared__ float red[K_];
  __shared__ float ginv;
  const int n = blockIdx.x, tid = threadIdx.x;

  if (tid < K_) {
    float s = 0.f;
#pragma unroll
    for (int ct = 0; ct < 32; ++ct) s += normpart[((size_t)(n * K_ + tid)) * 32 + ct];
    inv1[tid] = 1.f / fmaxf(sqrtf(s), 1e-12f);
    red[tid]  = s / fmaxf(s, 1e-24f);            // contribution to global sumsq
  }
  __syncthreads();
  if (tid == 0) {
    float g = 0.f;
#pragma unroll
    for (int k = 0; k < K_; ++k) g += red[k];
    ginv = 1.f / fmaxf(sqrtf(g), 1e-12f);
  }
  __syncthreads();

  const size_t base = (size_t)n * K_ * C_;
  const float gi = ginv;
  for (int e = tid; e < K_ * C_; e += 256)
    out[base + e] = vlad[base + e] * inv1[e >> 9] * gi;   // e/512 = k
}

// ---------------------------------------------------------------------------
// Workspace layout (bytes), total ~72.4 MB:
//   xn_pm  f16 [N][P][C]      0         33554432
//   xn_cm  f16 [N][C][P]      33554432  33554432
//   sa     f16 [N][K][P]      67108864   4194304
//   wh     f16 [K][C]         71303168     65536
//   vlad   f32 [N][K][C]      71368704   4194304
//   sasum_part f32 [N][8][K]  75563008     65536
//   sasum  f32 [N][K]         75628544      8192
//   normpart f32 [N][K][32]   75636736    262144
// ---------------------------------------------------------------------------
extern "C" void kernel_launch(void* const* d_in, const int* in_sizes, int n_in,
                              void* d_out, int out_size, void* d_ws, size_t ws_size,
                              hipStream_t stream) {
  const float* x      = (const float*)d_in[0];
  const float* conv_w = (const float*)d_in[1];
  const float* conv_b = (const float*)d_in[2];
  const float* cent   = (const float*)d_in[3];
  float* out = (float*)d_out;

  char* ws = (char*)d_ws;
  _Float16* xn_pm = (_Float16*)(ws + 0);
  _Float16* xn_cm = (_Float16*)(ws + 33554432);
  _Float16* sa    = (_Float16*)(ws + 67108864);
  _Float16* wh    = (_Float16*)(ws + 71303168);
  float* vlad     = (float*)(ws + 71368704);
  float* sasum_p  = (float*)(ws + 75563008);
  float* sasum    = (float*)(ws + 75628544);
  float* normp    = (float*)(ws + 75636736);

  normalize_kernel<<<128, 256, 0, stream>>>(x, conv_w, xn_cm, xn_pm, wh);
  logits_softmax_kernel<<<256, 256, 0, stream>>>(xn_pm, wh, conv_b, sa, sasum_p);
  reduce_sasum_kernel<<<N_, K_, 0, stream>>>(sasum_p, sasum);
  vlad_kernel<<<512, 256, 0, stream>>>(sa, xn_cm, sasum, cent, vlad, normp);
  finalize_kernel<<<N_, 256, 0, stream>>>(vlad, normp, out);
}
